// Attention_3513283248742
// MI455X (gfx1250) — matrix operand order repaired and checked by reading, not compiled, not run on hardware
//
#include <hip/hip_runtime.h>

// ---------------------------------------------------------------------------
// Bilinear attention for MI455X (gfx1250, wave32, WMMA 16x16x32 f16 -> f32)
//   b   : [32, 1024, 1024] f32   (passages)
//   h   : [32,  256, 1024] f32   (queries)
//   W_b : [ 1, 1024, 1024] f32   (bilinear weight)
//   out : [32,  256, 1024] f32   (context)
// Pipeline: Wt=f16(W^T) -> hW=f16(h@W) -> fused scores/softmax/ctx
// ---------------------------------------------------------------------------

typedef __attribute__((ext_vector_type(16))) _Float16 v16h;
typedef __attribute__((ext_vector_type(8)))  float    v8f;

#define D 1024

__device__ inline v8f wmma_f16(v16h a, v16h b, v8f c) {
    // D = A(16x32 f16) * B(32x16 f16) + C(16x16 f32)
    return __builtin_amdgcn_wmma_f32_16x16x32_f16(
        /*neg_a=*/false, a, /*neg_b=*/false, b,
        /*c_mod=*/(short)0, c, /*reuse_a=*/false, /*reuse_b=*/false);
}

// ---------------------------------------------------------------------------
// Kernel 1: Wt[e][d] = (f16) W[d][e]   (LDS-tiled transpose, 32x32 tiles)
// ---------------------------------------------------------------------------
__global__ __launch_bounds__(256) void transposeW(const float* __restrict__ W,
                                                  _Float16* __restrict__ Wt) {
    __shared__ float tile[32][33];
    const int ebase = blockIdx.x * 32;
    const int dbase = blockIdx.y * 32;
    const int tx = threadIdx.x & 31;
    const int ty = threadIdx.x >> 5;   // 0..7
#pragma unroll
    for (int j = 0; j < 32; j += 8)
        tile[ty + j][tx] = W[(size_t)(dbase + ty + j) * D + ebase + tx];
    __syncthreads();
#pragma unroll
    for (int j = 0; j < 32; j += 8)
        Wt[(size_t)(ebase + ty + j) * D + dbase + tx] = (_Float16)tile[tx][ty + j];
}

// ---------------------------------------------------------------------------
// Kernel 2: hW[row][e] = f16( sum_d h[row][d] * W[d][e] ),  rows = B*TH = 8192
// One WG = 16 rows x 1024 cols. 8 waves x 8 col-tiles, K-loop of 32.
// Per K-step: preload all 8 B-frags (clause of b128 loads), then WMMA chain.
// ---------------------------------------------------------------------------
__global__ __launch_bounds__(256) void gemm_hW(const float* __restrict__ h,
                                               const _Float16* __restrict__ Wt,
                                               _Float16* __restrict__ hW) {
    const int lane = threadIdx.x & 31;
    const int wave = threadIdx.x >> 5;
    const int c  = lane & 15;          // A-row / B-col within tile
    const int hi = lane >> 4;          // half-wave selector
    const int row_base = blockIdx.x * 16;

    v8f acc[8] = {};
    const float* hrow = h + (size_t)(row_base + c) * D;

    for (int kb = 0; kb < D; kb += 32) {
        // A fragment (f32 -> f16 inline, contiguous)
        v16h a;
        const float* p0 = hrow + kb + hi * 16;
#pragma unroll
        for (int i = 0; i < 8; ++i) {
            a[i]     = (_Float16)p0[i];        // K = hi*16 + i
            a[8 + i] = (_Float16)p0[8 + i];    // K = hi*16 + 8 + i
        }
        // Preload all 8 B fragments, then run the WMMA chain.
        v16h bf[8];
#pragma unroll
        for (int j = 0; j < 8; ++j) {
            const _Float16* q = Wt + (size_t)((wave * 8 + j) * 16 + c) * D + kb + hi * 16;
#pragma unroll
            for (int i = 0; i < 16; ++i) bf[j][i] = q[i];   // K contiguous, 32B
        }
#pragma unroll
        for (int j = 0; j < 8; ++j) acc[j] = wmma_f16(a, bf[j], acc[j]);
    }
#pragma unroll
    for (int j = 0; j < 8; ++j) {
        const int n0 = (wave * 8 + j) * 16 + c;
#pragma unroll
        for (int v = 0; v < 8; ++v)
            hW[(size_t)(row_base + v + hi * 8) * D + n0] = (_Float16)acc[j][v];
    }
}

// ---------------------------------------------------------------------------
// Kernel 3: fused  S = hW @ b^T  ->  softmax  ->  ctx = P @ b
// One WG = one (batch, 16-query tile). 256 threads = 8 waves.
// LDS: pBuf 32KB (f16 logits -> unnormalized exp = A operand of ctx GEMM),
//      bTile double-buffered 2x10KB (32k x 128d transposed chunks of b),
//      1/rowSum folded into the final store.
// ---------------------------------------------------------------------------
__global__ __launch_bounds__(256) void attn_kernel(const float* __restrict__ b,
                                                   const _Float16* __restrict__ hW,
                                                   float* __restrict__ out) {
    __shared__ _Float16 pBuf[16 * 1024];        // 32 KB
    __shared__ _Float16 bTile[2][128 * 40];     // 2 x 10 KB (stride 40 -> 16B aligned)
    __shared__ float    rowSum[16];

    const int lane  = threadIdx.x & 31;
    const int wave  = threadIdx.x >> 5;
    const int c     = lane & 15;
    const int hi    = lane >> 4;
    const int qt    = blockIdx.x;               // 0..15
    const int batch = blockIdx.y;               // 0..31
    const int qrow  = batch * 256 + qt * 16;    // flat row into hW / out
    const float* bB = b + (size_t)batch * D * D;

    // ---------------- Stage A: scores -> pBuf (f16 logits) ----------------
    {
        v8f acc[8] = {};
        const _Float16* arow = hW + (size_t)(qrow + c) * D;
        for (int eb = 0; eb < D; eb += 32) {
            v16h a;
            const _Float16* p0 = arow + eb + hi * 16;
#pragma unroll
            for (int i = 0; i < 8; ++i) { a[i] = p0[i]; a[8 + i] = p0[8 + i]; }
            // Preload all 8 B fragments (f32 rows of b, 64B each, cvt inline)
            v16h bf[8];
#pragma unroll
            for (int j = 0; j < 8; ++j) {
                const float* q = bB + (size_t)((wave * 8 + j) * 16 + c) * D + eb + hi * 16;
#pragma unroll
                for (int i = 0; i < 16; ++i) bf[j][i] = (_Float16)q[i];
            }
#pragma unroll
            for (int j = 0; j < 8; ++j) acc[j] = wmma_f16(a, bf[j], acc[j]);
        }
#pragma unroll
        for (int j = 0; j < 8; ++j) {
            const int n0 = (wave * 8 + j) * 16 + c;
#pragma unroll
            for (int v = 0; v < 8; ++v)
                pBuf[(v + hi * 8) * 1024 + n0] = (_Float16)acc[j][v];
        }
    }
    __syncthreads();

    // ---------------- Stage B: softmax (store exp(s-max), keep row sums) ---
    {
        const int row = threadIdx.x >> 4;   // 16 threads per row
        const int s   = threadIdx.x & 15;
        _Float16* prow = pBuf + row * 1024;
        float m = -3.0e38f;
#pragma unroll 4
        for (int i = 0; i < 64; ++i) m = fmaxf(m, (float)prow[s + 16 * i]);
#pragma unroll
        for (int off = 8; off > 0; off >>= 1) m = fmaxf(m, __shfl_xor(m, off, 16));
        float lsum = 0.0f;
#pragma unroll 4
        for (int i = 0; i < 64; ++i) {
            float e = __expf((float)prow[s + 16 * i] - m);
            prow[s + 16 * i] = (_Float16)e;   // unnormalized P, in [0,1]
            lsum += e;
        }
#pragma unroll
        for (int off = 8; off > 0; off >>= 1) lsum += __shfl_xor(lsum, off, 16);
        if (s == 0) rowSum[row] = lsum;
    }
    __syncthreads();

    float rs[8];
#pragma unroll
    for (int v = 0; v < 8; ++v) rs[v] = 1.0f / rowSum[v + hi * 8];

    // ------- Stage C: ctx = P @ b. 8 d-chunks x 32 k-steps, flattened ------
    // Double-buffered transpose staging: each step stages the NEXT step's
    // 32k x 128d chunk of b while the WMMA for the current step executes.
    {
        v8f accC;
#pragma unroll
        for (int v = 0; v < 8; ++v) accC[v] = 0.0f;

        // prologue: stage step 0 (chunk 0, k-rows 0..31) into buffer 0
        {
            const float* src = bB;
#pragma unroll
            for (int i = 0; i < 16; ++i) {
                const int idx = threadIdx.x + 256 * i;
                const int k = idx >> 7, dd = idx & 127;
                bTile[0][dd * 40 + k] = (_Float16)src[(size_t)k * D + dd];
            }
        }

        for (int step = 0; step < 256; ++step) {
            const int cur = step & 1;
            __syncthreads();   // staging of buf[cur] done; buf[cur^1] free

            if (step + 1 < 256) {   // stage next step into the other buffer
                const int s1 = step + 1;
                const float* src = bB + (size_t)((s1 & 31) * 32) * D + (s1 >> 5) * 128;
#pragma unroll
                for (int i = 0; i < 16; ++i) {
                    const int idx = threadIdx.x + 256 * i;
                    const int k = idx >> 7, dd = idx & 127;
                    bTile[cur ^ 1][dd * 40 + k] = (_Float16)src[(size_t)k * D + dd];
                }
            }
            if (step + 2 < 256) {   // pull step+2's lines toward GL2
                const int s2 = step + 2;
                __builtin_prefetch(bB + (size_t)((s2 & 31) * 32) * D + (s2 >> 5) * 128
                                       + (threadIdx.x & 127), 0, 1);
            }

            const int kb = (step & 31) * 32;
            // A-frag from pBuf (f16, contiguous in k)
            v16h a;
            const _Float16* p0 = pBuf + c * 1024 + kb + hi * 16;
#pragma unroll
            for (int i = 0; i < 8; ++i) { a[i] = p0[i]; a[8 + i] = p0[8 + i]; }
            // B-frag from staged transposed chunk (aligned ds_load_b128)
            v16h bfv;
            const _Float16* q = &bTile[cur][(wave * 16 + c) * 40 + hi * 16];
#pragma unroll
            for (int i = 0; i < 16; ++i) bfv[i] = q[i];

            accC = wmma_f16(a, bfv, accC);

            if ((step & 31) == 31) {   // chunk finished: store + reset
                const int dcol = (step >> 5) * 128 + wave * 16 + c;
#pragma unroll
                for (int v = 0; v < 8; ++v) {
                    out[(size_t)(qrow + v + hi * 8) * D + dcol] = accC[v] * rs[v];
                    accC[v] = 0.0f;
                }
            }
        }
    }
}

// ---------------------------------------------------------------------------
extern "C" void kernel_launch(void* const* d_in, const int* in_sizes, int n_in,
                              void* d_out, int out_size, void* d_ws, size_t ws_size,
                              hipStream_t stream) {
    const float* b = (const float*)d_in[0];   // [32,1024,1024]
    const float* h = (const float*)d_in[1];   // [32, 256,1024]
    const float* W = (const float*)d_in[2];   // [ 1,1024,1024]
    float* out = (float*)d_out;               // [32, 256,1024]

    _Float16* Wt = (_Float16*)d_ws;           // 1024*1024 f16 (2 MB)
    _Float16* hW = Wt + (size_t)D * D;        // 8192*1024 f16 (16.8 MB)

    transposeW<<<dim3(32, 32), 256, 0, stream>>>(W, Wt);
    gemm_hW  <<<dim3(512),    256, 0, stream>>>(h, Wt, hW);
    attn_kernel<<<dim3(16, 32), 256, 0, stream>>>(b, hW, out);
}
